// Quantization_25108378812572
// MI455X (gfx1250) — matrix operand order repaired and checked by reading, not compiled, not run on hardware
//
#include <hip/hip_runtime.h>
#include <hip/hip_bf16.h>

#define N_ROWS 32768
#define HDIM   1024
#define CB     8
#define DDIM   128
#define KCODE  512

typedef _Float16     v16h  __attribute__((ext_vector_type(16)));
typedef float        v8f   __attribute__((ext_vector_type(8)));
typedef unsigned int u32x4 __attribute__((ext_vector_type(4)));
typedef float        f32x4 __attribute__((ext_vector_type(4)));

union AFrag { v16h v; _Float16 e[16]; };
union BFrag { v16h v; u32x4 u[2]; };

// ---------------------------------------------------------------------------
// Prep: codesT32[c][k][d] = codes[c][d][k]  (exact fp32, d-contiguous)
//       codesT16[c][k][d] = (f16)(codes[c][d][k] / ||codes[c][:][k]||)
// One wave per (c,k).
// ---------------------------------------------------------------------------
__global__ __launch_bounds__(256) void vq_prep(const float* __restrict__ codes,
                                               _Float16* __restrict__ codesT16,
                                               float* __restrict__ codesT32) {
  int gw   = (blockIdx.x * 256 + threadIdx.x) >> 5;   // global wave id
  int lane = threadIdx.x & 31;
  int c = gw >> 9;          // /512
  int k = gw & 511;
  float v[4];
  float ss = 0.f;
#pragma unroll
  for (int i = 0; i < 4; ++i) {
    int d = lane + 32 * i;
    v[i] = codes[((size_t)(c * DDIM + d)) * KCODE + k];
    ss += v[i] * v[i];
  }
#pragma unroll
  for (int off = 16; off >= 1; off >>= 1) ss += __shfl_xor(ss, off, 32);
  float rn = rsqrtf(ss);
#pragma unroll
  for (int i = 0; i < 4; ++i) {
    int d = lane + 32 * i;
    size_t o = ((size_t)(c * KCODE + k)) * DDIM + d;
    codesT32[o] = v[i];
    codesT16[o] = (_Float16)(v[i] * rn);
  }
}

// ---------------------------------------------------------------------------
// Main: block = 16 rows; wave w handles codebook c = w.
// WMMA 16x16x32 f16 logits -> + gumbel -> argmax -> gather nh + loss partial.
// ---------------------------------------------------------------------------
__global__ __launch_bounds__(256) void vq_main(const float* __restrict__ hid,
                                               const float* __restrict__ gum,
                                               const _Float16* __restrict__ codesT16,
                                               const float* __restrict__ codesT32,
                                               float* __restrict__ out,
                                               float* __restrict__ partials) {
  const int c    = threadIdx.x >> 5;    // wave id == codebook index
  const int lane = threadIdx.x & 31;
  const int half = lane >> 4;
  const int mlo  = lane & 15;
  const int rowbase = blockIdx.x * 16;

  // ---- A fragments: 16 rows x 128 d, f16 (contiguous k = 16*half + s), + row sumsq
  AFrag a[4];
  float ssq = 0.f;
  const float* hrow = hid + (size_t)(rowbase + mlo) * HDIM + c * DDIM;
#pragma unroll
  for (int ch = 0; ch < 4; ++ch) {
    int kb = ch * 32 + 16 * half;
#pragma unroll
    for (int j = 0; j < 8; ++j) {
      float f0 = hrow[kb + j];
      float f1 = hrow[kb + 8 + j];
      ssq += f0 * f0 + f1 * f1;
      a[ch].e[j]     = (_Float16)f0;
      a[ch].e[8 + j] = (_Float16)f1;
    }
  }
  float tot  = ssq + __shfl_xor(ssq, 16, 32);
  float rinv = rsqrtf(tot);             // lane l: 1/||h[rowbase+(l&15), c, :]||
  // accumulator row m = r + 8*half -> pull rinv from lane holding that row
  float rv[8];
#pragma unroll
  for (int r = 0; r < 8; ++r) rv[r] = __shfl(rinv, r + 8 * half, 32);

  const float* gptr[8];
#pragma unroll
  for (int r = 0; r < 8; ++r)
    gptr[r] = gum + (size_t)(rowbase + r + 8 * half) * (CB * KCODE) + c * KCODE + mlo;

  float bestv[8];
  int   besti[8];
#pragma unroll
  for (int r = 0; r < 8; ++r) { bestv[r] = -__builtin_inff(); besti[r] = 0; }

  // ---- 32 code tiles of 16 columns, K-dim 128 = 4 x wmma_f32_16x16x32_f16
  for (int kt = 0; kt < 32; ++kt) {
    v8f acc = {0.f, 0.f, 0.f, 0.f, 0.f, 0.f, 0.f, 0.f};
    const int col = kt * 16 + mlo;
    const _Float16* bbase = codesT16 + (size_t)(c * KCODE + col) * DDIM + 16 * half;
#pragma unroll
    for (int ch = 0; ch < 4; ++ch) {
      BFrag b;
      const u32x4* bp = (const u32x4*)(bbase + ch * 32);
      b.u[0] = bp[0];
      b.u[1] = bp[1];
      acc = __builtin_amdgcn_wmma_f32_16x16x32_f16(false, a[ch].v, false, b.v,
                                                   (short)0, acc, false, false);
    }
#pragma unroll
    for (int r = 0; r < 8; ++r) {
      float sc = acc[r] * rv[r] + gptr[r][kt * 16];
      if (sc > bestv[r]) { bestv[r] = sc; besti[r] = col; }  // first-max kept
    }
  }

  // ---- cross-lane argmax within each 16-lane group (same row set)
#pragma unroll
  for (int r = 0; r < 8; ++r) {
#pragma unroll
    for (int off = 1; off < 16; off <<= 1) {
      float ov = __shfl_xor(bestv[r], off, 32);
      int   oi = __shfl_xor(besti[r], off, 32);
      if (ov > bestv[r] || (ov == bestv[r] && oi < besti[r])) {
        bestv[r] = ov; besti[r] = oi;
      }
    }
  }

  // ---- gather exact fp32 code rows -> nh, accumulate ||nh-h|| loss partial
  float lossAcc = 0.f;
  for (int m = 0; m < 16; ++m) {
    int idx = __shfl(besti[m & 7], (m < 8) ? 0 : 16, 32);
    const f32x4 cv = *(const f32x4*)(codesT32 + (size_t)(c * KCODE + idx) * DDIM + lane * 4);
    const f32x4 hv = *(const f32x4*)(hid + (size_t)(rowbase + m) * HDIM + c * DDIM + lane * 4);
    float d0 = cv[0] - hv[0], d1 = cv[1] - hv[1];
    float d2 = cv[2] - hv[2], d3 = cv[3] - hv[3];
    float ss = d0 * d0 + d1 * d1 + d2 * d2 + d3 * d3;
    *(f32x4*)(out + (size_t)(rowbase + m) * HDIM + c * DDIM + lane * 4) = cv;
#pragma unroll
    for (int off = 16; off >= 1; off >>= 1) ss += __shfl_xor(ss, off, 32);
    if (lane == 0) lossAcc += sqrtf(ss);
  }
  if (lane == 0) partials[(size_t)blockIdx.x * CB + c] = lossAcc;
}

// ---------------------------------------------------------------------------
// Deterministic final loss reduction (fixed order every call).
// ---------------------------------------------------------------------------
__global__ __launch_bounds__(256) void vq_lossred(const float* __restrict__ partials,
                                                  float* __restrict__ lossOut, int n) {
  __shared__ float s[256];
  float a = 0.f;
  for (int i = threadIdx.x; i < n; i += 256) a += partials[i];
  s[threadIdx.x] = a;
  __syncthreads();
  for (int st = 128; st > 0; st >>= 1) {
    if ((int)threadIdx.x < st) s[threadIdx.x] += s[threadIdx.x + st];
    __syncthreads();
  }
  if (threadIdx.x == 0) lossOut[0] = 1.2f * s[0];  // (BETA=0.2 + ALPHA=1.0)
}

extern "C" void kernel_launch(void* const* d_in, const int* in_sizes, int n_in,
                              void* d_out, int out_size, void* d_ws, size_t ws_size,
                              hipStream_t stream) {
  const float* hid   = (const float*)d_in[0];   // [32768, 1024]
  const float* codes = (const float*)d_in[1];   // [8, 128, 512]
  const float* gum   = (const float*)d_in[2];   // [32768, 8, 512]
  float* out = (float*)d_out;                   // [32768*1024] nh + [1] loss

  _Float16* codesT16 = (_Float16*)d_ws;                          // 1 MB
  float*    codesT32 = (float*)((char*)d_ws + (1u << 20));       // 2 MB
  float*    partials = (float*)((char*)d_ws + (3u << 20));       // 64 KB

  vq_prep<<<512, 256, 0, stream>>>(codes, codesT16, codesT32);
  vq_main<<<N_ROWS / 16, 256, 0, stream>>>(hid, gum, codesT16, codesT32, out, partials);
  vq_lossred<<<1, 256, 0, stream>>>(partials, out + (size_t)N_ROWS * HDIM,
                                    (N_ROWS / 16) * CB);
  (void)in_sizes; (void)n_in; (void)out_size; (void)ws_size;
}
